// CausalAttention_56856777065046
// MI455X (gfx1250) — matrix operand order repaired and checked by reading, not compiled, not run on hardware
//
#include <hip/hip_runtime.h>
#include <hip/hip_bf16.h>
#include <math.h>

// Problem constants (from reference): B=2, T=2048, HID=1024, H=16, D=64
#define Bc   2
#define Tc   2048
#define HIDc 1024
#define Hc   16
#define Dc   64

typedef __bf16 bf16;
typedef __attribute__((ext_vector_type(16))) __bf16 v16bf;
typedef __attribute__((ext_vector_type(8)))  float  v8f;

union BF2U { bf16 b; unsigned short u; };
union U8V  { uint4 v; bf16 h[8]; };

__device__ __forceinline__ U8V ld8(const bf16* p) { U8V u; u.v = *(const uint4*)p; return u; }

__device__ __forceinline__ v8f wmma_bf(v16bf a, v16bf b, v8f c) {
  // D(f32 16x16) = A(bf16 16x32) * B(bf16 32x16) + C
  return __builtin_amdgcn_wmma_f32_16x16x32_bf16(false, a, false, b, (short)0, c, false, false);
}

// ---------------- fp32 -> bf16 conversion kernels ----------------
__global__ void k_f2bf(const float* __restrict__ in, bf16* __restrict__ out, int n) {
  int i = blockIdx.x * blockDim.x + threadIdx.x;
  if (i < n) out[i] = (bf16)in[i];
}

// 1024x1024 transpose-convert: out[n*K+k] = in[k*N+n]  (so WMMA B-tiles load contiguously)
__global__ void k_f2bf_tr(const float* __restrict__ in, bf16* __restrict__ out) {
  int i = blockIdx.x * blockDim.x + threadIdx.x;
  if (i >= HIDc * HIDc) return;
  int k = i >> 10, n = i & 1023;
  out[(size_t)n * HIDc + k] = (bf16)in[i];
}

// ---------------- WMMA GEMM: [4096 x 1024] (bf16) @ Bt[N,K] (bf16) ----------------
// Shared 16x16-tile WMMA body; three epilogue variants below.
#define GEMM_BODY(ACC)                                                              \
  const int K = HIDc;                                                               \
  const int lane = threadIdx.x & 31;                                                \
  const int tile = blockIdx.x * 4 + (threadIdx.x >> 5);                             \
  const int m0 = (tile >> 6) * 16;                                                  \
  const int n0 = (tile & 63) * 16;                                                  \
  const int mA = lane & 15, kbA = (lane >> 4) * 16;                                 \
  const int koB = (lane >> 4) * 16;                                                 \
  const bf16* arow = A + (size_t)(m0 + mA) * K;                                     \
  const bf16* brow = Bt + (size_t)(n0 + (lane & 15)) * K;                           \
  v8f ACC = {};                                                                     \
  for (int k0 = 0; k0 < K; k0 += 32) {                                              \
    v16bf a, bm;                                                                    \
    U8V a0 = ld8(arow + k0 + kbA);                                                  \
    U8V a1 = ld8(arow + k0 + kbA + 8);                                              \
    U8V b0 = ld8(brow + k0 + koB);                                                  \
    U8V b1 = ld8(brow + k0 + koB + 8);                                              \
    _Pragma("unroll")                                                               \
    for (int i = 0; i < 8; ++i) {                                                   \
      a[i] = a0.h[i]; a[8 + i] = a1.h[i];                                           \
      bm[i] = b0.h[i]; bm[8 + i] = b1.h[i];                                         \
    }                                                                               \
    ACC = wmma_bf(a, bm, ACC);                                                      \
  }                                                                                 \
  const int baseM = (lane >> 4) * 8;                                                \
  const int n = n0 + (lane & 15);

// Q/K variant: output scattered to [B, H, T, D] bf16
__global__ __launch_bounds__(128) void k_gemm_qk(
    const bf16* __restrict__ A, const bf16* __restrict__ Bt, bf16* __restrict__ Out) {
  GEMM_BODY(acc)
  const int h = n >> 6, d = n & 63;
#pragma unroll
  for (int r = 0; r < 8; ++r) {
    const int m = m0 + baseM + r;
    const int b = m / Tc, t = m % Tc;
    Out[((size_t)(b * Hc + h) * Tc + t) * Dc + d] = (bf16)acc[r];
  }
}

// V variant: output scattered TRANSPOSED to [B, H, D, T] bf16 so the attention
// kernel's P@V B-operand loads become contiguous 128-bit loads along t.
__global__ __launch_bounds__(128) void k_gemm_vT(
    const bf16* __restrict__ A, const bf16* __restrict__ Bt, bf16* __restrict__ Out) {
  GEMM_BODY(acc)
  const int h = n >> 6, d = n & 63;
#pragma unroll
  for (int r = 0; r < 8; ++r) {
    const int m = m0 + baseM + r;
    const int b = m / Tc, t = m % Tc;
    Out[((size_t)(b * Hc + h) * Dc + d) * Tc + t] = (bf16)acc[r];
  }
}

// Output-projection variant: plain row-major fp32 output
__global__ __launch_bounds__(128) void k_gemm_out(
    const bf16* __restrict__ A, const bf16* __restrict__ Bt, float* __restrict__ Out) {
  GEMM_BODY(acc)
#pragma unroll
  for (int r = 0; r < 8; ++r) {
    const int m = m0 + baseM + r;
    Out[(size_t)m * HIDc + n] = acc[r];
  }
}

// ---------------- fused attention: raw + score-norm + causal online softmax + P@V ----------------
// One wave32 per (b, h, 16-row tile). EXEC is all-ones everywhere WMMA executes.
__global__ __launch_bounds__(32) void k_attn(
    const bf16* __restrict__ Qb, const bf16* __restrict__ Kb, const bf16* __restrict__ Vt,
    const float* __restrict__ s_prev, const float* __restrict__ alpha,
    const float* __restrict__ gamma_p, const float* __restrict__ beta_p,
    float* __restrict__ raw_out, bf16* __restrict__ ctxb) {
  __shared__ float mu_s[16];
  __shared__ float sdi_s[16];
  __shared__ unsigned short p_lds[16 * 32];

  const int lane = threadIdx.x;
  const int ntT = Tc / 16;
  const int t0 = (blockIdx.x % ntT) * 16;
  const int bh = blockIdx.x / ntT;
  const int h = bh % Hc;
  const int b = bh / Hc;

  const bf16* Qh = Qb + (size_t)bh * Tc * Dc;
  const bf16* Kh = Kb + (size_t)bh * Tc * Dc;
  const bf16* Vh = Vt + (size_t)bh * Dc * Tc;   // [D, T] per (b,h)
  const float* Sh = s_prev + (size_t)bh * Tc * Tc;
  float* Rh = raw_out + (size_t)bh * Tc * Tc;

  // ---- per-row mean / population-std of s_prev (full row, pre-mask) ----
  {
    const int row = lane & 15, half = lane >> 4; // 2 lanes per row, 1024 elems each
    const float4* p = (const float4*)(Sh + (size_t)(t0 + row) * Tc + half * (Tc / 2));
    float s = 0.f, ss = 0.f;
    for (int j = 0; j < Tc / 8; ++j) {
      float4 v = p[j];
      s += v.x + v.y + v.z + v.w;
      ss += v.x * v.x + v.y * v.y + v.z * v.z + v.w * v.w;
    }
    s  += __shfl_xor(s, 16, 32);
    ss += __shfl_xor(ss, 16, 32);
    if (lane < 16) {
      float mu = s * (1.f / Tc);
      float var = ss * (1.f / Tc) - mu * mu;
      mu_s[row] = mu;
      sdi_s[row] = 1.f / (sqrtf(fmaxf(var, 0.f)) + 1e-5f);
    }
  }
  __asm__ volatile("s_wait_dscnt 0" ::: "memory");

  const float gate = 1.f / (1.f + __expf(-alpha[h]));
  const float scl = gamma_p[h];
  const float shf = beta_p[h];
  const float invsq = 0.125f; // 1/sqrt(64)

  // Q tiles in WMMA A-layout (d chunks 0..31 and 32..63)
  const int mA = lane & 15;
  const int kbA = (lane >> 4) * 16;
  v16bf qa0, qa1;
  {
    const bf16* qrow = Qh + (size_t)(t0 + mA) * Dc;
    U8V a0 = ld8(qrow + kbA);
    U8V a1 = ld8(qrow + kbA + 8);
    U8V a2 = ld8(qrow + 32 + kbA);
    U8V a3 = ld8(qrow + 32 + kbA + 8);
#pragma unroll
    for (int i = 0; i < 8; ++i) { qa0[i] = a0.h[i]; qa0[8 + i] = a1.h[i]; qa1[i] = a2.h[i]; qa1[8 + i] = a3.h[i]; }
  }

  const int baseM = (lane >> 4) * 8; // C-layout row base for this lane
  const int nC = lane & 15;          // C-layout column / B-layout N
  const int koB = (lane >> 4) * 16;  // B-layout K offset

  float mu_r[8], sdi_r[8];
#pragma unroll
  for (int r = 0; r < 8; ++r) { mu_r[r] = mu_s[baseM + r]; sdi_r[r] = sdi_s[baseM + r]; }

  v8f zero = {};
  v8f oacc[4] = { zero, zero, zero, zero };
  float rmax[8], lsum[8];
#pragma unroll
  for (int r = 0; r < 8; ++r) { rmax[r] = -3.0e38f; lsum[r] = 0.f; }

  for (int s0 = 0; s0 < Tc; s0 += 32) {
    // prefetch next s_prev tile rows into cache (global_prefetch_b8); lanes 0-15 = rows
    if (s0 + 32 < Tc)
      __builtin_prefetch(Sh + (size_t)(t0 + (lane & 15)) * Tc + s0 + 32, 0, 0);

    // ---- scores S = Q @ K^T for columns [s0, s0+32) : 4 WMMAs ----
    v8f sc[2];
#pragma unroll
    for (int c = 0; c < 2; ++c) {
      const bf16* krow = Kh + (size_t)(s0 + c * 16 + nC) * Dc;
      v16bf kb0, kb1;
      U8V x0 = ld8(krow + koB);
      U8V x1 = ld8(krow + koB + 8);
      U8V x2 = ld8(krow + 32 + koB);
      U8V x3 = ld8(krow + 32 + koB + 8);
#pragma unroll
      for (int i = 0; i < 8; ++i) { kb0[i] = x0.h[i]; kb0[8 + i] = x1.h[i]; kb1[i] = x2.h[i]; kb1[8 + i] = x3.h[i]; }
      v8f t = {};
      t = wmma_bf(qa0, kb0, t);
      t = wmma_bf(qa1, kb1, t);
      sc[c] = t;
    }

    // ---- raw = S/sqrt(D) + gate*(scale*norm(s_prev)+shift); write FULL (unmasked) raw ----
    float rawv0[8], rawv1[8];
#pragma unroll
    for (int c = 0; c < 2; ++c) {
      const int col = s0 + c * 16 + nC;
      const float* sp = Sh + col;
      float* rp = Rh + col;
#pragma unroll
      for (int r = 0; r < 8; ++r) {
        const int row = t0 + baseM + r;
        const float spv = sp[(size_t)row * Tc];
        const float v = sc[c][r] * invsq + gate * (scl * (spv - mu_r[r]) * sdi_r[r] + shf);
        if (c == 0) rawv0[r] = v; else rawv1[r] = v;
        rp[(size_t)row * Tc] = v;
      }
    }

    // ---- causal online softmax + P@V (only tiles touching the causal region) ----
    if (s0 <= t0 + 15) {
      float pm0[8], pm1[8], tmax[8];
#pragma unroll
      for (int r = 0; r < 8; ++r) {
        const int row = t0 + baseM + r;
        pm0[r] = (s0 + nC <= row) ? rawv0[r] : -3.0e38f;
        pm1[r] = (s0 + 16 + nC <= row) ? rawv1[r] : -3.0e38f;
        tmax[r] = fmaxf(pm0[r], pm1[r]);
      }
#pragma unroll
      for (int off = 1; off < 16; off <<= 1)
#pragma unroll
        for (int r = 0; r < 8; ++r) tmax[r] = fmaxf(tmax[r], __shfl_xor(tmax[r], off, 32));

      float corr[8];
#pragma unroll
      for (int r = 0; r < 8; ++r) {
        const float nm = fmaxf(rmax[r], tmax[r]);
        corr[r] = __expf(rmax[r] - nm);
        rmax[r] = nm;
        pm0[r] = (pm0[r] > -1.0e37f) ? __expf(pm0[r] - nm) : 0.f;
        pm1[r] = (pm1[r] > -1.0e37f) ? __expf(pm1[r] - nm) : 0.f;
      }
      float psum[8];
#pragma unroll
      for (int r = 0; r < 8; ++r) psum[r] = pm0[r] + pm1[r];
#pragma unroll
      for (int off = 1; off < 16; off <<= 1)
#pragma unroll
        for (int r = 0; r < 8; ++r) psum[r] += __shfl_xor(psum[r], off, 32);
#pragma unroll
      for (int r = 0; r < 8; ++r) lsum[r] = lsum[r] * corr[r] + psum[r];

      // stage P (C-layout) to LDS as bf16, reload in A-layout for P@V
#pragma unroll
      for (int r = 0; r < 8; ++r) {
        BF2U c0, c1;
        c0.b = (bf16)pm0[r];
        c1.b = (bf16)pm1[r];
        p_lds[(baseM + r) * 32 + nC] = c0.u;
        p_lds[(baseM + r) * 32 + 16 + nC] = c1.u;
      }
      __asm__ volatile("s_wait_dscnt 0" ::: "memory");
      v16bf pa;
#pragma unroll
      for (int i = 0; i < 8; ++i) {
        BF2U lo, hi;
        lo.u = p_lds[mA * 32 + kbA + i];
        hi.u = p_lds[mA * 32 + kbA + 8 + i];
        pa[i] = lo.b;
        pa[8 + i] = hi.b;
      }
      // V is [D, T]: each lane's B-operand slice is 16 contiguous bf16 along t.
      // (s0 <= 2016 here and tiles are 32-wide with T%32==0, so t-indices stay < T.)
#pragma unroll
      for (int dsub = 0; dsub < 4; ++dsub) {
#pragma unroll
        for (int r = 0; r < 8; ++r) oacc[dsub][r] *= corr[r];
        const bf16* vrow = Vh + (size_t)(dsub * 16 + nC) * Tc + s0 + koB;
        U8V y0 = ld8(vrow);
        U8V y1 = ld8(vrow + 8);
        v16bf vb;
#pragma unroll
        for (int i = 0; i < 8; ++i) { vb[i] = y0.h[i]; vb[8 + i] = y1.h[i]; }
        oacc[dsub] = wmma_bf(pa, vb, oacc[dsub]);
      }
    }
  }

  // ---- finalize: ctx = O / lsum, merged layout [B, T, H*D] as bf16 ----
#pragma unroll
  for (int r = 0; r < 8; ++r) {
    const int t = t0 + baseM + r;
    const float inv = 1.f / lsum[r];
    bf16* crow = ctxb + ((size_t)b * Tc + t) * (Hc * Dc) + h * Dc;
#pragma unroll
    for (int dsub = 0; dsub < 4; ++dsub)
      crow[dsub * 16 + nC] = (bf16)(oacc[dsub][r] * inv);
  }
}

extern "C" void kernel_launch(void* const* d_in, const int* in_sizes, int n_in,
                              void* d_out, int out_size, void* d_ws, size_t ws_size,
                              hipStream_t stream) {
  (void)in_sizes; (void)n_in; (void)out_size; (void)ws_size;
  const float* x      = (const float*)d_in[0];
  const float* s_prev = (const float*)d_in[1];
  const float* Wq     = (const float*)d_in[2];
  const float* Wk     = (const float*)d_in[3];
  const float* Wv     = (const float*)d_in[4];
  const float* Wo     = (const float*)d_in[5];
  const float* alpha  = (const float*)d_in[6];
  const float* gma    = (const float*)d_in[7];
  const float* bta    = (const float*)d_in[8];

  float* out = (float*)d_out;                       // [B, T, HID] fp32
  float* raw = out + (size_t)Bc * Tc * HIDc;        // [B, H, T, T] fp32

  // workspace (bf16): x, 4 transposed weights, Q/K/Vt, ctx  => 48 MB total
  bf16* ws = (bf16*)d_ws;
  const size_t NX = (size_t)Bc * Tc * HIDc; // 4 Mi elems
  const size_t NW = (size_t)HIDc * HIDc;    // 1 Mi elems
  bf16* xb   = ws;
  bf16* WqT  = xb + NX;
  bf16* WkT  = WqT + NW;
  bf16* WvT  = WkT + NW;
  bf16* WoT  = WvT + NW;
  bf16* Qb   = WoT + NW;
  bf16* Kb   = Qb + NX;
  bf16* Vt   = Kb + NX;    // [B, H, D, T]
  bf16* ctxb = Vt + NX;

  k_f2bf<<<(int)(NX / 256), 256, 0, stream>>>(x, xb, (int)NX);
  k_f2bf_tr<<<(int)(NW / 256), 256, 0, stream>>>(Wq, WqT);
  k_f2bf_tr<<<(int)(NW / 256), 256, 0, stream>>>(Wk, WkT);
  k_f2bf_tr<<<(int)(NW / 256), 256, 0, stream>>>(Wv, WvT);
  k_f2bf_tr<<<(int)(NW / 256), 256, 0, stream>>>(Wo, WoT);

  const int gemmBlocks = (256 * 64) / 4; // 16384 tiles, 4 waves/block
  k_gemm_qk<<<gemmBlocks, 128, 0, stream>>>(xb, WqT, Qb);
  k_gemm_qk<<<gemmBlocks, 128, 0, stream>>>(xb, WkT, Kb);
  k_gemm_vT<<<gemmBlocks, 128, 0, stream>>>(xb, WvT, Vt);

  k_attn<<<Bc * Hc * (Tc / 16), 32, 0, stream>>>(Qb, Kb, Vt, s_prev, alpha, gma, bta, raw, ctxb);

  k_gemm_out<<<gemmBlocks, 128, 0, stream>>>(ctxb, WoT, out);
}
